// SVDAttention_59425167508018
// MI455X (gfx1250) — matrix operand order repaired and checked by reading, not compiled, not run on hardware
//
#include <hip/hip_runtime.h>
#include <hip/hip_bf16.h>
#include <math.h>

// ---------------------------------------------------------------------------
// SVD-attention with Jacobi graph filter, MI455X (gfx1250, wave32, WMMA).
// Memory-bound (~34 GFLOP vs ~1.3 GB traffic @ 23.3 TB/s): f16 intermediates,
// f16 WMMA / f32 accumulate, register-double-buffered LDS tiles (1 barrier
// per K-step), pipelined B fragments, packed-b32 transpose staging.
// ---------------------------------------------------------------------------

typedef _Float16 half_t;
typedef __attribute__((ext_vector_type(2)))  _Float16 h2;
typedef __attribute__((ext_vector_type(4)))  _Float16 h4;
typedef __attribute__((ext_vector_type(8)))  _Float16 h8;
typedef __attribute__((ext_vector_type(16))) _Float16 v16h;
typedef __attribute__((ext_vector_type(8)))  float    v8f;

#define NB 8
#define NH 8
#define SEQ 4096
#define RNK 128
#define NBH 64
#define LDT 40      // padded LDS row (32 data halves + 8 pad) -> conflict-free frags
#define LDB5 136    // padded LDS row for 128-wide B tile

#define WMMA_F16(A, Bv, C) \
  __builtin_amdgcn_wmma_f32_16x16x32_f16(false, (A), false, (Bv), (short)0, (C), false, false)

// ---- WMMA fragment loaders (layouts per CDNA5 ISA 7.12.2) -----------------
__device__ __forceinline__ v16h frag2(const half_t* p0, const half_t* p1) {
  h8 a = *(const h8*)p0;
  h8 b = *(const h8*)p1;
  v16h f;
#pragma unroll
  for (int i = 0; i < 8; ++i) { f[i] = a[i]; f[i + 8] = b[i]; }
  return f;
}
// A 16x32: lane m=lane&15, hf=lane>>4; chunks k=[8hf..8hf+7], [16+8hf..23+8hf].
__device__ __forceinline__ v16h load_frag_A(const half_t* tile, int ld) {
  int lane = threadIdx.x & 31;
  const half_t* row = tile + (lane & 15) * ld;
  int hf = lane >> 4;
  return frag2(row + 8 * hf, row + 16 + 8 * hf);
}
// B 32x16 staged [n][k]: lane n=lane&15, kh=lane>>4; chunks k=[8kh..8kh+7], [16+8kh..23+8kh].
__device__ __forceinline__ v16h load_frag_B(const half_t* tile, int ld) {
  int lane = threadIdx.x & 31;
  const half_t* p = tile + (lane & 15) * ld + 8 * (lane >> 4);
  return frag2(p, p + 16);
}

// ---- Jacobi filter: elementwise degree-4 recurrence (a=b=1, [-1,1]) -------
__device__ __forceinline__ float jacobi_eval(float sig, const float* al, const float* cb) {
  const float a = 1.f, bp = 1.f, lo = -1.f, ro = 1.f;
  float x_pp = 1.f;
  float acc = cb[0];
  float c1 = (a - bp) * 0.5f - (a + bp + 2.f) * 0.5f * (lo + ro) / (ro - lo);
  float c2 = (a + bp + 2.f) / (ro - lo);
  float x_p = al[0] * (c1 + c2 * sig);
  acc += cb[1] * x_p;
#pragma unroll
  for (int Lk = 2; Lk <= 4; ++Lk) {
    float fL = (float)Lk;
    float coef_l     = 2.f * fL * (fL + a + bp) * (2.f * fL - 2.f + a + bp);
    float coef_lm1_1 = (2.f * fL + a + bp - 1.f) * (2.f * fL + a + bp) * (2.f * fL + a + bp - 2.f);
    float coef_lm1_2 = (2.f * fL + a + bp - 1.f) * (a * a - bp * bp);
    float coef_lm2   = 2.f * (fL - 1.f + a) * (fL - 1.f + bp) * (2.f * fL + a + bp);
    float tmp1 = al[Lk - 1] * (coef_lm1_1 / coef_l);
    float tmp2 = al[Lk - 1] * (coef_lm1_2 / coef_l);
    float tmp3 = al[Lk - 1] * al[Lk - 2] * (coef_lm2 / coef_l);
    float tmp1_2 = tmp1 * (2.f / (ro - lo));
    float tmp2_2 = tmp1 * ((ro + lo) / (ro - lo)) + tmp2;
    float nx = tmp1_2 * sig * x_p - tmp2_2 * x_p - tmp3 * x_pp;
    acc += cb[Lk] * nx;
    x_pp = x_p; x_p = nx;
  }
  return acc;
}

// ---- K1: fused softmax(U over r) + sigmoid + Jacobi filter ----------------
__global__ __launch_bounds__(256) void k_softmax_u(
    const float* __restrict__ U, const float* __restrict__ Sg,
    const float* __restrict__ araw, const float* __restrict__ comb,
    half_t* __restrict__ Ut, half_t* __restrict__ Uf) {
  int wave = (blockIdx.x << 3) + (threadIdx.x >> 5);  // (b*L + l)*H + h
  int lane = threadIdx.x & 31;
  float al[4], cb[5];
#pragma unroll
  for (int k = 0; k < 4; ++k) al[k] = tanhf(araw[k]);  // BASE_ALPHA = 1
#pragma unroll
  for (int k = 0; k < 5; ++k) cb[k] = comb[k];

  size_t base = (size_t)wave * RNK + lane * 4;
  float4 u = *(const float4*)(U + base);
  float m = fmaxf(fmaxf(u.x, u.y), fmaxf(u.z, u.w));
#pragma unroll
  for (int off = 16; off; off >>= 1) m = fmaxf(m, __shfl_xor(m, off, 32));
  float e0 = expf(u.x - m), e1 = expf(u.y - m), e2 = expf(u.z - m), e3 = expf(u.w - m);
  float s = e0 + e1 + e2 + e3;
#pragma unroll
  for (int off = 16; off; off >>= 1) s += __shfl_xor(s, off, 32);
  float inv = 1.f / s;

  float4 sg = *(const float4*)(Sg + base);
  float f0 = jacobi_eval(1.f / (1.f + expf(-sg.x)), al, cb);
  float f1 = jacobi_eval(1.f / (1.f + expf(-sg.y)), al, cb);
  float f2 = jacobi_eval(1.f / (1.f + expf(-sg.z)), al, cb);
  float f3 = jacobi_eval(1.f / (1.f + expf(-sg.w)), al, cb);

  int h = wave & 7;
  int bl = wave >> 3;
  int l = bl & (SEQ - 1);
  int b = bl >> 12;
  size_t drow = ((size_t)(b * NH + h) * SEQ + l) * RNK + lane * 4;
  h4 ut = { (half_t)(e0 * inv), (half_t)(e1 * inv), (half_t)(e2 * inv), (half_t)(e3 * inv) };
  h4 uf = { (half_t)(e0 * inv * f0), (half_t)(e1 * inv * f1),
            (half_t)(e2 * inv * f2), (half_t)(e3 * inv * f3) };
  *(h4*)(Ut + drow) = ut;
  *(h4*)(Uf + drow) = uf;
}

// ---- K2: softmax(V over sequence) -> Vt f16 in [b,h,r,s] ------------------
__global__ __launch_bounds__(256) void k_softmax_v(
    const float* __restrict__ V, half_t* __restrict__ Vt) {
  __shared__ float red[256];
  int row = blockIdx.x;            // (b*H + h)*128 + r
  int b = row >> 10;
  int h = (row >> 7) & 7;
  int r = row & 127;
  int t = threadIdx.x;
  size_t base = (size_t)b * (SEQ * NH * RNK) + (size_t)h * RNK + r;
  float v[16];
  float m = -INFINITY;
#pragma unroll
  for (int i = 0; i < 16; ++i) {
    v[i] = V[base + (size_t)(t + 256 * i) * (NH * RNK)];
    m = fmaxf(m, v[i]);
  }
  red[t] = m; __syncthreads();
#pragma unroll
  for (int off = 128; off > 0; off >>= 1) {
    if (t < off) red[t] = fmaxf(red[t], red[t + off]);
    __syncthreads();
  }
  m = red[0]; __syncthreads();
  float s = 0.f;
#pragma unroll
  for (int i = 0; i < 16; ++i) { v[i] = expf(v[i] - m); s += v[i]; }
  red[t] = s; __syncthreads();
#pragma unroll
  for (int off = 128; off > 0; off >>= 1) {
    if (t < off) red[t] += red[t + off];
    __syncthreads();
  }
  float inv = 1.f / red[0];
  half_t* dst = Vt + (size_t)row * SEQ;
#pragma unroll
  for (int i = 0; i < 16; ++i) dst[t + 256 * i] = (half_t)(v[i] * inv);
}

// ---- K3: values [b,l,h,d] f32 -> valsT [b,h,s,d] f16 ----------------------
__global__ __launch_bounds__(256) void k_vals_convert(
    const float* __restrict__ vals, half_t* __restrict__ valsT) {
  size_t tid = (size_t)blockIdx.x * 256 + threadIdx.x;  // over float4 elements
  int d4 = tid & 31;
  size_t rest = tid >> 5;                 // (b*L + s)*H + h
  int h = rest & 7;
  size_t bs = rest >> 3;
  int s = bs & (SEQ - 1);
  int b = (int)(bs >> 12);
  float4 v = *(const float4*)(vals + tid * 4);
  h4 o = { (half_t)v.x, (half_t)v.y, (half_t)v.z, (half_t)v.w };
  size_t dst = ((size_t)(b * NH + h) * SEQ + s) * RNK + d4 * 4;
  *(h4*)(valsT + dst) = o;
}

// ---- K4: kv[r,d] = sum_s Vt[r,s]*vals[s,d]; 128x128x4096 per (b,h) --------
// register double-buffered staging; result stored transposed kvT[d][r].
__global__ __launch_bounds__(256) void k_kv_gemm(
    const half_t* __restrict__ Vt, const half_t* __restrict__ valsT,
    half_t* __restrict__ kvT) {
  __shared__ half_t At[2][128 * LDT];
  __shared__ half_t Bt[2][128 * LDT];
  int bh = blockIdx.x;
  int t = threadIdx.x, w = t >> 5, lane = t & 31;
  const half_t* Asrc = Vt + (size_t)bh * RNK * SEQ;
  const half_t* Bsrc = valsT + (size_t)bh * SEQ * RNK;

  h8 ra0, ra1;           // staged A rows (k contiguous)
  h2 rb0[4], rb1[4];     // staged B rows for the packed transpose
  const int srow = t >> 1, sko = (t & 1) * 16;
  const int sn0 = (t & 63) * 2, skb = (t >> 6);

  auto sload = [&](int kb) {
    const half_t* s = Asrc + (size_t)srow * SEQ + kb + sko;
    ra0 = *(const h8*)s; ra1 = *(const h8*)(s + 8);
#pragma unroll
    for (int it = 0; it < 4; ++it) {
      int kk = (skb + it * 4) * 2;
      rb0[it] = *(const h2*)(Bsrc + (size_t)(kb + kk) * RNK + sn0);
      rb1[it] = *(const h2*)(Bsrc + (size_t)(kb + kk + 1) * RNK + sn0);
    }
  };
  auto sstore = [&](int buf) {
    *(h8*)&At[buf][srow * LDT + sko]     = ra0;
    *(h8*)&At[buf][srow * LDT + sko + 8] = ra1;
#pragma unroll
    for (int it = 0; it < 4; ++it) {
      int kk = (skb + it * 4) * 2;
      h2 c0 = { rb0[it].x, rb1[it].x };
      h2 c1 = { rb0[it].y, rb1[it].y };
      *(h2*)&Bt[buf][sn0 * LDT + kk]       = c0;
      *(h2*)&Bt[buf][(sn0 + 1) * LDT + kk] = c1;
    }
  };

  const v8f zero = {0.f, 0.f, 0.f, 0.f, 0.f, 0.f, 0.f, 0.f};
  v8f acc[8];
#pragma unroll
  for (int i = 0; i < 8; ++i) acc[i] = zero;

  sload(0);
  sstore(0);
  int buf = 0;
  for (int kb = 0; kb < SEQ; kb += 32, buf ^= 1) {
    bool more = (kb + 32) < SEQ;
    if (more) sload(kb + 32);           // global loads fly over the barrier+MMAs
    __syncthreads();                    // current buffer visible to all waves
    v16h a = load_frag_A(&At[buf][16 * w * LDT], LDT);
    v16h bcur = load_frag_B(&Bt[buf][0], LDT);
#pragma unroll
    for (int nt = 0; nt < 8; ++nt) {
      v16h bnext;
      if (nt < 7) bnext = load_frag_B(&Bt[buf][16 * (nt + 1) * LDT], LDT);
      acc[nt] = WMMA_F16(a, bcur, acc[nt]);
      if (nt < 7) bcur = bnext;
    }
    if (more) sstore(buf ^ 1);          // dump regs into ping-pong buffer
  }
  int M0 = 16 * w + 8 * (lane >> 4);
  int N0 = lane & 15;
#pragma unroll
  for (int nt = 0; nt < 8; ++nt) {
    int d = 16 * nt + N0;
    h8 o;
#pragma unroll
    for (int v = 0; v < 8; ++v) o[v] = (half_t)acc[nt][v];
    *(h8*)&kvT[((size_t)bh * RNK + d) * RNK + M0] = o;   // packed b128 store
  }
}

// ---- K5: out[l,d] = sum_r Uf[l,r]*kv[r,d]; 4096x128x128 per (b,h) ---------
// B (kvT, [d][r]) resident in LDS; A fragments straight from global (K=128,
// each A row consumed by exactly one wave) -> zero per-K-step barriers.
__global__ __launch_bounds__(256) void k_out_gemm(
    const half_t* __restrict__ Uf, const half_t* __restrict__ kvT,
    float* __restrict__ out) {
  __shared__ half_t Bt[128 * LDB5];
  int bh = blockIdx.x >> 5, mb = blockIdx.x & 31;
  int b = bh >> 3, h = bh & 7;
  int t = threadIdx.x, w = t >> 5, lane = t & 31;
  {  // stage full 128x128 B once
    int n = t >> 1, off = (t & 1) * 64;
    const half_t* s = kvT + (size_t)bh * (RNK * RNK) + n * RNK + off;
#pragma unroll
    for (int j = 0; j < 8; ++j)
      *(h8*)&Bt[n * LDB5 + off + 8 * j] = *(const h8*)(s + 8 * j);
  }
  const v8f zero = {0.f, 0.f, 0.f, 0.f, 0.f, 0.f, 0.f, 0.f};
  v8f acc[8];
#pragma unroll
  for (int i = 0; i < 8; ++i) acc[i] = zero;
  const half_t* Arow = Uf + ((size_t)bh * SEQ + mb * 128 + 16 * w) * RNK;
  __syncthreads();
#pragma unroll
  for (int kb = 0; kb < RNK; kb += 32) {
    v16h a = load_frag_A(Arow + kb, RNK);      // direct global b128 x2
    v16h bcur = load_frag_B(Bt + kb, LDB5);
#pragma unroll
    for (int nt = 0; nt < 8; ++nt) {
      v16h bnext;
      if (nt < 7) bnext = load_frag_B(Bt + 16 * (nt + 1) * LDB5 + kb, LDB5);
      acc[nt] = WMMA_F16(a, bcur, acc[nt]);
      if (nt < 7) bcur = bnext;
    }
  }
  int M0 = 16 * w + 8 * (lane >> 4);
  int N0 = lane & 15;
#pragma unroll
  for (int nt = 0; nt < 8; ++nt) {
    int d = 16 * nt + N0;
#pragma unroll
    for (int v = 0; v < 8; ++v) {
      int l = mb * 128 + M0 + v;
      out[(((size_t)b * SEQ + l) * NH + h) * RNK + d] = acc[nt][v];
    }
  }
}

// ---- K6/K7: ortho losses |X^T X - I|; one LDS tile serves A and B ---------
// TR=true : X is [SEQ][RNK] (Ut) -> packed-b32 transpose staging.
// TR=false: X is [RNK][SEQ] (Vt) -> direct b128 staging.
template <bool TR>
__global__ __launch_bounds__(256) void k_sym_loss(
    const half_t* __restrict__ X, float* __restrict__ loss) {
  __shared__ half_t Tt[2][128 * LDT];
  __shared__ float red[256];
  int bh = blockIdx.x;
  int b = bh >> 3;
  int t = threadIdx.x, w = t >> 5, lane = t & 31;
  const half_t* base = X + (size_t)bh * SEQ * RNK;

  h8 ra0, ra1;
  h2 rb0[4], rb1[4];
  const int srow = t >> 1, sko = (t & 1) * 16;
  const int sn0 = (t & 63) * 2, skb = (t >> 6);

  auto sload = [&](int kb) {
    if (TR) {
#pragma unroll
      for (int it = 0; it < 4; ++it) {
        int kk = (skb + it * 4) * 2;
        rb0[it] = *(const h2*)(base + (size_t)(kb + kk) * RNK + sn0);
        rb1[it] = *(const h2*)(base + (size_t)(kb + kk + 1) * RNK + sn0);
      }
    } else {
      const half_t* s = base + (size_t)srow * SEQ + kb + sko;
      ra0 = *(const h8*)s; ra1 = *(const h8*)(s + 8);
    }
  };
  auto sstore = [&](int buf) {
    if (TR) {
#pragma unroll
      for (int it = 0; it < 4; ++it) {
        int kk = (skb + it * 4) * 2;
        h2 c0 = { rb0[it].x, rb1[it].x };
        h2 c1 = { rb0[it].y, rb1[it].y };
        *(h2*)&Tt[buf][sn0 * LDT + kk]       = c0;
        *(h2*)&Tt[buf][(sn0 + 1) * LDT + kk] = c1;
      }
    } else {
      *(h8*)&Tt[buf][srow * LDT + sko]     = ra0;
      *(h8*)&Tt[buf][srow * LDT + sko + 8] = ra1;
    }
  };

  const v8f zero = {0.f, 0.f, 0.f, 0.f, 0.f, 0.f, 0.f, 0.f};
  v8f acc[8];
#pragma unroll
  for (int i = 0; i < 8; ++i) acc[i] = zero;

  sload(0);
  sstore(0);
  int buf = 0;
  for (int kb = 0; kb < SEQ; kb += 32, buf ^= 1) {
    bool more = (kb + 32) < SEQ;
    if (more) sload(kb + 32);
    __syncthreads();
    v16h a = load_frag_A(&Tt[buf][16 * w * LDT], LDT);
    v16h bcur = load_frag_B(&Tt[buf][0], LDT);
#pragma unroll
    for (int nt = 0; nt < 8; ++nt) {
      v16h bnext;
      if (nt < 7) bnext = load_frag_B(&Tt[buf][16 * (nt + 1) * LDT], LDT);
      acc[nt] = WMMA_F16(a, bcur, acc[nt]);
      if (nt < 7) bcur = bnext;
    }
    if (more) sstore(buf ^ 1);
  }
  float part = 0.f;
  int M0 = 16 * w + 8 * (lane >> 4);
  int N0 = lane & 15;
#pragma unroll
  for (int nt = 0; nt < 8; ++nt) {
#pragma unroll
    for (int v = 0; v < 8; ++v) {
      int rg = M0 + v, sg = 16 * nt + N0;
      float val = acc[nt][v] - (rg == sg ? 1.f : 0.f);
      part += fabsf(val);
    }
  }
  red[t] = part; __syncthreads();
#pragma unroll
  for (int off = 128; off > 0; off >>= 1) {
    if (t < off) red[t] += red[t + off];
    __syncthreads();
  }
  if (t == 0) atomicAdd(loss + b, red[0] * (1.f / (NH * 128.f * 128.f)));
}

__global__ void k_zero_loss(float* loss) {
  if (threadIdx.x < NB) loss[threadIdx.x] = 0.f;
}

// ---------------------------------------------------------------------------
extern "C" void kernel_launch(void* const* d_in, const int* in_sizes, int n_in,
                              void* d_out, int out_size, void* d_ws, size_t ws_size,
                              hipStream_t stream) {
  const float* U     = (const float*)d_in[0];
  const float* Sigma = (const float*)d_in[1];
  const float* V     = (const float*)d_in[2];
  const float* vals  = (const float*)d_in[3];
  const float* araw  = (const float*)d_in[4];
  const float* comb  = (const float*)d_in[5];

  float* out  = (float*)d_out;
  float* loss = out + (size_t)NB * SEQ * NH * RNK;

  const size_t TENS = (size_t)NB * SEQ * NH * RNK;  // 33,554,432 elements
  char* ws = (char*)d_ws;
  half_t* Ut    = (half_t*)(ws);
  half_t* Uf    = (half_t*)(ws + TENS * 2);
  half_t* Vt    = (half_t*)(ws + TENS * 4);
  half_t* valsT = (half_t*)(ws + TENS * 6);
  half_t* kvT   = (half_t*)(ws + TENS * 8);

  k_zero_loss<<<1, 32, 0, stream>>>(loss);
  k_softmax_u<<<(NB * SEQ * NH) / 8, 256, 0, stream>>>(U, Sigma, araw, comb, Ut, Uf);
  k_softmax_v<<<NB * NH * RNK, 256, 0, stream>>>(V, Vt);
  k_vals_convert<<<(int)(TENS / 4 / 256), 256, 0, stream>>>(vals, valsT);
  k_kv_gemm<<<NBH, 256, 0, stream>>>(Vt, valsT, kvT);
  k_out_gemm<<<NBH * 32, 256, 0, stream>>>(Uf, kvT, out);
  k_sym_loss<true><<<NBH, 256, 0, stream>>>(Ut, loss);   // |Ut^T Ut - I|
  k_sym_loss<false><<<NBH, 256, 0, stream>>>(Vt, loss);  // |Vt Vt^T - I|
}
